// ESN_40046275068581
// MI455X (gfx1250) — matrix operand order repaired and checked by reading, not compiled, not run on hardware
//
#include <hip/hip_runtime.h>
#include <hip/hip_bf16.h>
#include <math.h>

// ---------------------------------------------------------------------------
// ESN on MI455X (gfx1250):
//   kernel 1: zero per-timestep barrier counters (re-run every launch)
//   kernel 2: persistent recurrence. 64 WGs, each holds a 32x2048 bf16 slice
//             of W in LDS (128 KB of the 320 KB WGP LDS). Per step:
//             x_{t-1} staged global->LDS via GLOBAL_LOAD_ASYNC_TO_LDS_B128
//             (ASYNCcnt), matvec via v_wmma_f32_16x16x32_bf16 with x
//             replicated across B columns (C-chained accumulation, no
//             N-reduction), + W_in projection in VALU, tanh, store bf16
//             state, then a device-wide barrier (one counter per step).
//   kernel 3: outputs = states(bf16) @ W_out^T + b via the same WMMA.
// ---------------------------------------------------------------------------

typedef __bf16 bf16;
typedef __attribute__((ext_vector_type(16))) __bf16 v16bf;
typedef __attribute__((ext_vector_type(8)))  __bf16 v8bf;
typedef __attribute__((ext_vector_type(4)))  __bf16 v4bf;
typedef __attribute__((ext_vector_type(8)))  float  v8f;
typedef __attribute__((ext_vector_type(4)))  float  v4f;

#define T_STEPS   8192
#define DIN       64
#define N_RES     2048
#define DOUT      64
#define NWG       64
#define ROWS_WG   32            // N_RES / NWG
#define TPB       256           // 8 wave32
#define WL_STRIDE 2056          // 2048 + 8 bf16 pad -> row stride 4112 B (4 DW skew mod 64 banks)

// LDS layout (bytes)
#define OFF_WL    0
#define OFF_WIN   (OFF_WL  + ROWS_WG * WL_STRIDE * 2)   // 131584
#define OFF_XS    (OFF_WIN + ROWS_WG * DIN * 4)         // +8192
#define OFF_US    (OFF_XS  + N_RES * 2)                 // +4096
#define OFF_RED   (OFF_US  + DIN * 4)                   // +256
#define SMEM_BYTES (OFF_RED + 8 * 16 * 4)               // +512 -> 144640

// generic -> LDS(AS3) offset for async-to-LDS destination VGPR
__device__ __forceinline__ unsigned lds_offset_of(const void* p) {
    return (unsigned)(uintptr_t)(__attribute__((address_space(3))) const void*)p;
}

__global__ void esn_init_kernel(unsigned* __restrict__ cnt) {
    int g = blockIdx.x * blockDim.x + threadIdx.x;
    if (g < T_STEPS) cnt[g] = 0u;
}

__global__ void __launch_bounds__(TPB, 1)
esn_recurrence_kernel(const float* __restrict__ u,
                      const float* __restrict__ W_in,
                      const float* __restrict__ W,
                      bf16*        __restrict__ states,   // [T_STEPS][N_RES] bf16
                      unsigned*    __restrict__ cnt)      // [T_STEPS]
{
    extern __shared__ char smem[];
    bf16*  Wl   = (bf16*) (smem + OFF_WL);
    float* Winl = (float*)(smem + OFF_WIN);
    bf16*  xs   = (bf16*) (smem + OFF_XS);
    float* us   = (float*)(smem + OFF_US);
    float* red  = (float*)(smem + OFF_RED);

    const int tid  = threadIdx.x;
    const int lane = tid & 31;
    const int wave = tid >> 5;
    const int wg   = blockIdx.x;
    const int row0 = wg * ROWS_WG;

    const int rt  = wave & 1;          // row-tile (0..1) this wave owns
    const int q   = wave >> 1;         // K-quarter (0..3)
    const int m   = lane & 15;
    const int hi  = lane >> 4;         // 0 / 1 : selects K-half per WMMA layout
    const int klo = hi * 8;

    // ---- one-time preload: W slice (fp32 -> bf16) into LDS ----
    for (int g4 = tid; g4 < ROWS_WG * N_RES / 4; g4 += TPB) {
        int e = g4 * 4;
        int r = e / N_RES, c = e % N_RES;
        v4f w4 = *(const v4f*)(W + (size_t)(row0 + r) * N_RES + c);
        v4bf p;
        p[0] = (bf16)w4[0]; p[1] = (bf16)w4[1];
        p[2] = (bf16)w4[2]; p[3] = (bf16)w4[3];
        *(v4bf*)(Wl + r * WL_STRIDE + c) = p;
    }
    // W_in slice (row-major contiguous for this WG's rows)
    for (int i = tid; i < ROWS_WG * DIN; i += TPB)
        Winl[i] = W_in[(size_t)row0 * DIN + i];
    __syncthreads();

    const bf16*    arow   = Wl + (rt * 16 + m) * WL_STRIDE;
    const unsigned xs_off = lds_offset_of(xs + tid * 8);   // 16 B per thread

    for (int t = 0; t < T_STEPS; ++t) {
        // ---- stage x_{t-1}: async global->LDS (no VGPR round-trip) ----
        if (t > 0) {
            unsigned long long gaddr =
                (unsigned long long)(const void*)(states + (size_t)(t - 1) * N_RES + tid * 8);
            asm volatile("global_load_async_to_lds_b128 %0, %1, off"
                         :: "v"(xs_off), "v"(gaddr) : "memory");
        }
        if (tid < DIN) us[tid] = u[(size_t)t * DIN + tid];
        if (tid == 0 && t + 1 < T_STEPS)
            __builtin_prefetch(u + (size_t)(t + 1) * DIN, 0, 0);   // global_prefetch
        asm volatile("s_wait_asynccnt 0x0" ::: "memory");
        __syncthreads();

        // ---- matvec partials via WMMA: D[m,n] identical over n (B replicated) ----
        v8f acc = {0.f, 0.f, 0.f, 0.f, 0.f, 0.f, 0.f, 0.f};
        if (t > 0) {
            #pragma unroll
            for (int kc = 0; kc < 16; ++kc) {
                const int kb = q * 512 + kc * 32;
                v16bf a;
                *((v8bf*)&a)     = *(const v8bf*)(arow + kb + hi * 16);        // K 0-7 / 16-23
                *((v8bf*)&a + 1) = *(const v8bf*)(arow + kb + hi * 16 + 8);   // K 8-15 / 24-31
                v16bf b = *(const v16bf*)(xs + kb + hi * 16);              // broadcast read
                acc = __builtin_amdgcn_wmma_f32_16x16x32_bf16(
                          false, a, false, b, (short)0, acc, false, false);
            }
        }
        // lanes 0 / 16 of each wave hold rows rt*16+{0..7} / {8..15}
        if (m == 0) {
            #pragma unroll
            for (int i = 0; i < 8; ++i) red[wave * 16 + hi * 8 + i] = acc[i];
        }
        __syncthreads();

        // ---- combine K-quarters, add input projection, tanh, emit state ----
        if (tid < ROWS_WG) {
            const int rrt = tid >> 4, rl = tid & 15;
            float s = 0.f;
            #pragma unroll
            for (int qq = 0; qq < 4; ++qq) s += red[(qq * 2 + rrt) * 16 + rl];
            float p = 0.f;
            #pragma unroll
            for (int j = 0; j < DIN; ++j) p += Winl[tid * DIN + j] * us[j];
            float x = tanhf(p + s);
            states[(size_t)t * N_RES + row0 + tid] = (bf16)x;
        }

        // ---- device-wide barrier: one fresh counter per timestep ----
        __threadfence();
        __syncthreads();
        if (tid == 0) {
            __hip_atomic_fetch_add(&cnt[t], 1u, __ATOMIC_RELEASE, __HIP_MEMORY_SCOPE_AGENT);
            while (__hip_atomic_load(&cnt[t], __ATOMIC_ACQUIRE, __HIP_MEMORY_SCOPE_AGENT) < NWG)
                __builtin_amdgcn_s_sleep(1);
        }
        __syncthreads();
        __threadfence();
    }
}

__global__ void __launch_bounds__(TPB)
esn_output_kernel(const bf16*  __restrict__ states,   // [T][N_RES] bf16
                  const float* __restrict__ W_out,    // [DOUT][N_RES] f32
                  const float* __restrict__ b_out,    // [DOUT]
                  float*       __restrict__ out)      // [T][DOUT]
{
    const int tid  = threadIdx.x;
    const int lane = tid & 31;
    const int wave = tid >> 5;
    const int wid  = blockIdx.x * (TPB / 32) + wave;  // 0..2047 tiles
    const int t0   = (wid >> 2) * 16;
    const int o0   = (wid & 3) * 16;
    const int m    = lane & 15;
    const int hi   = lane >> 4;
    const int klo  = hi * 8;

    const bf16*  arow = states + (size_t)(t0 + m) * N_RES;
    const float* brow = W_out + (size_t)(o0 + m) * N_RES;

    v8f acc = {0.f, 0.f, 0.f, 0.f, 0.f, 0.f, 0.f, 0.f};
    for (int kb = 0; kb < N_RES; kb += 32) {
        v16bf a;
        *((v8bf*)&a)     = *(const v8bf*)(arow + kb + hi * 16);
        *((v8bf*)&a + 1) = *(const v8bf*)(arow + kb + hi * 16 + 8);
        // 16 consecutive f32 of W_out row -> 4x float4 loads, cvt to bf16
        const float* bp = brow + kb + hi * 16;
        v16bf b;
        #pragma unroll
        for (int g4 = 0; g4 < 4; ++g4) {
            v4f w4 = *(const v4f*)(bp + g4 * 4);
            b[g4 * 4 + 0] = (bf16)w4[0];
            b[g4 * 4 + 1] = (bf16)w4[1];
            b[g4 * 4 + 2] = (bf16)w4[2];
            b[g4 * 4 + 3] = (bf16)w4[3];
        }
        acc = __builtin_amdgcn_wmma_f32_16x16x32_bf16(
                  false, a, false, b, (short)0, acc, false, false);
    }
    const float bo = b_out[o0 + m];
    #pragma unroll
    for (int i = 0; i < 8; ++i) {
        const int row = t0 + hi * 8 + i;              // C/D: VGPR i -> M = i (+8 for hi half)
        out[(size_t)row * DOUT + o0 + m] = acc[i] + bo;
    }
}

extern "C" void kernel_launch(void* const* d_in, const int* in_sizes, int n_in,
                              void* d_out, int out_size, void* d_ws, size_t ws_size,
                              hipStream_t stream) {
    const float* u     = (const float*)d_in[0];   // (8192, 64)
    const float* W_in  = (const float*)d_in[1];   // (2048, 64)
    const float* W     = (const float*)d_in[2];   // (2048, 2048)
    const float* W_out = (const float*)d_in[3];   // (64, 2048)
    const float* b_out = (const float*)d_in[4];   // (64,)
    float* out = (float*)d_out;

    // workspace: [0, 32KB) barrier counters; [32KB, +32MB) bf16 states
    unsigned* cnt    = (unsigned*)d_ws;
    bf16*     states = (bf16*)((char*)d_ws + 32768);

    esn_init_kernel<<<T_STEPS / TPB, TPB, 0, stream>>>(cnt);
    esn_recurrence_kernel<<<NWG, TPB, SMEM_BYTES, stream>>>(u, W_in, W, states, cnt);
    const int out_blocks = (T_STEPS / 16) * (DOUT / 16) / (TPB / 32);  // 256
    esn_output_kernel<<<out_blocks, TPB, 0, stream>>>(states, W_out, b_out, out);
}
